// GNN_49503793053942
// MI455X (gfx1250) — matrix operand order repaired and checked by reading, not compiled, not run on hardware
//
#include <hip/hip_runtime.h>
#include <cfloat>

// ---------------------------------------------------------------------------
// Types for gfx1250 WMMA (wave32): A/B = 16 x f16 (8 VGPRs), C/D = 8 x f32
// ---------------------------------------------------------------------------
typedef __attribute__((ext_vector_type(16))) _Float16 v16h;
typedef __attribute__((ext_vector_type(8)))  _Float16 h8;
typedef __attribute__((ext_vector_type(8)))  float    v8f;

#define HID 48

// Float atomic max via signed/unsigned int monotonic trick (init = -FLT_MAX).
__device__ __forceinline__ void atomicMaxF(float* addr, float v) {
  if (v >= 0.f) atomicMax((int*)addr, __float_as_int(v));
  else          atomicMin((unsigned int*)addr, __float_as_uint(v));
}

__device__ __forceinline__ h8 h8zero() {
  h8 r;
#pragma unroll
  for (int i = 0; i < 8; ++i) r[i] = (_Float16)0.f;
  return r;
}

__device__ __forceinline__ h8 cvt8(float4 a, float4 b) {
  h8 r;
  r[0] = (_Float16)a.x; r[1] = (_Float16)a.y; r[2] = (_Float16)a.z; r[3] = (_Float16)a.w;
  r[4] = (_Float16)b.x; r[5] = (_Float16)b.y; r[6] = (_Float16)b.z; r[7] = (_Float16)b.w;
  return r;
}

__device__ __forceinline__ h8 cvt8s(float4 a, float4 b, float s) {
  h8 r;
  r[0] = (_Float16)(a.x * s); r[1] = (_Float16)(a.y * s);
  r[2] = (_Float16)(a.z * s); r[3] = (_Float16)(a.w * s);
  r[4] = (_Float16)(b.x * s); r[5] = (_Float16)(b.y * s);
  r[6] = (_Float16)(b.z * s); r[7] = (_Float16)(b.w * s);
  return r;
}

// ---------------------------------------------------------------------------
// WMMA fragment loaders from LDS tiles (wave32 layouts per CDNA5 ISA 7.12.2)
//   A (16xK f16): lanes 0-15: row M=lane, K = {0..7} U {16..23} (+ kt*32)
//                 lanes 16-31: row M=lane-16, K = {8..15} U {24..31}
//   B (Kx16 f16): lane n (0-15): col N=n, K = {0..7} U {16..23} (+ kt*32)
//                 lane n+16:     col N=n, K = {8..15} U {24..31}
// Strides are multiples of 8 elements -> 16B-aligned ds_load_b128.
// ---------------------------------------------------------------------------
__device__ __forceinline__ v16h loadA(const _Float16* __restrict__ T,
                                      int lane, int kt, int stride) {
  const int half = lane >> 4, r = lane & 15;
  const _Float16* base = T + r * stride + kt * 32 + half * 8;
  union { v16h v; h8 q[2]; } u;
  u.q[0] = *(const h8*)(base);        // K = base+0..7
  u.q[1] = *(const h8*)(base + 16);   // K = base+16..23
  return u.v;
}

__device__ __forceinline__ v16h loadB(const _Float16* __restrict__ Wt,
                                      int lane, int nt, int kt, int stride) {
  const int half = lane >> 4, n = (lane & 15) + nt * 16;
  const _Float16* base = Wt + n * stride + kt * 32 + half * 8;
  union { v16h v; h8 q[2]; } u;
  u.q[0] = *(const h8*)(base);
  u.q[1] = *(const h8*)(base + 16);
  return u.v;
}

__device__ __forceinline__ v8f wmma_f16(v16h a, v16h b, v8f c) {
  return __builtin_amdgcn_wmma_f32_16x16x32_f16(false, a, false, b,
                                                (short)0, c, false, false);
}

// Stage W0 (dout x DIN f32) -> LDS f16 padded to DINP, and W1 (48x48) -> 48x64.
template <int DIN, int DINP, int NTHR>
__device__ __forceinline__ void stage_weights(
    const float* __restrict__ W0, const float* __restrict__ B0,
    const float* __restrict__ W1, const float* __restrict__ B1,
    _Float16* sW0, _Float16* sW1, float* sB0, float* sB1, int tid) {
  constexpr int CPR = DINP / 8;  // chunks per row
  for (int ch = tid; ch < HID * CPR; ch += NTHR) {
    const int n = ch / CPR, k = (ch - n * CPR) * 8;
    h8 out;
    if (k < DIN) {
      const float4* p = (const float4*)(W0 + n * DIN + k);
      out = cvt8(p[0], p[1]);
    } else {
      out = h8zero();
    }
    *(h8*)(sW0 + n * DINP + k) = out;
  }
  for (int ch = tid; ch < HID * 8; ch += NTHR) {
    const int n = ch >> 3, k = (ch & 7) * 8;
    h8 out;
    if (k < HID) {
      const float4* p = (const float4*)(W1 + n * HID + k);
      out = cvt8(p[0], p[1]);
    } else {
      out = h8zero();
    }
    *(h8*)(sW1 + n * 64 + k) = out;
  }
  for (int i = tid; i < HID; i += NTHR) { sB0[i] = B0[i]; sB1[i] = B1[i]; }
}

// ---------------------------------------------------------------------------
// Utility kernels
// ---------------------------------------------------------------------------
__global__ void k_fill(float* __restrict__ p, float v, int n) {
  int i = blockIdx.x * 256 + threadIdx.x;
  if (i < n) p[i] = v;
}

__global__ void k_count(const int* __restrict__ col, float* __restrict__ cnt, int nE) {
  int i = blockIdx.x * 256 + threadIdx.x;
  if (i < nE) atomicAdd(&cnt[col[i]], 1.f);
}

// ---------------------------------------------------------------------------
// Layer-0 edge MLP: din = 3 (too skinny for WMMA) -> scalar VALU, 1 thread/edge.
// ---------------------------------------------------------------------------
__global__ __launch_bounds__(256) void k_edge0(
    const float* __restrict__ attr, const int* __restrict__ col,
    const float* __restrict__ W0, const float* __restrict__ B0,
    const float* __restrict__ W1, const float* __restrict__ B1,
    _Float16* __restrict__ Eout, float* __restrict__ S, float* __restrict__ MX,
    int nE) {
  __shared__ float sW0[HID * 3], sB0[HID], sW1[HID * HID], sB1[HID];
  const int tid = threadIdx.x;
  for (int i = tid; i < HID * 3;   i += 256) sW0[i] = W0[i];
  for (int i = tid; i < HID * HID; i += 256) sW1[i] = W1[i];
  if (tid < HID) { sB0[tid] = B0[tid]; sB1[tid] = B1[tid]; }
  __syncthreads();

  const int m = blockIdx.x * 256 + tid;
  if (m >= nE) return;
  const float a0 = attr[m * 3 + 0], a1 = attr[m * 3 + 1], a2 = attr[m * 3 + 2];
  float h[HID];
#pragma unroll
  for (int n = 0; n < HID; ++n) {
    float v = sW0[n * 3] * a0 + sW0[n * 3 + 1] * a1 + sW0[n * 3 + 2] * a2 + sB0[n];
    h[n] = v > 0.f ? v : 0.f;
  }
  const int ci = col[m];
#pragma unroll 4
  for (int n = 0; n < HID; ++n) {
    float acc = sB1[n];
#pragma unroll
    for (int k = 0; k < HID; ++k) acc += sW1[n * HID + k] * h[k];
    Eout[m * HID + n] = (_Float16)acc;
    atomicAdd(&S[ci * HID + n], acc);
    atomicMaxF(&MX[ci * HID + n], acc);
  }
}

// ---------------------------------------------------------------------------
// Fused MLP2 (din -> 48 -> relu -> 48) via WMMA, edge version.
// 128 threads = 4 waves; each wave owns one 16-edge tile.
// A tile built in LDS with b128 chunked gathers of [x[row] | x[col] | e_prev].
// ---------------------------------------------------------------------------
template <int DIN, int DINP>
__global__ __launch_bounds__(128) void k_edge_wmma(
    const float* __restrict__ X, const _Float16* __restrict__ Ein,
    const int* __restrict__ row, const int* __restrict__ col,
    const float* __restrict__ W0, const float* __restrict__ B0,
    const float* __restrict__ W1, const float* __restrict__ B1,
    _Float16* __restrict__ Eout, float* __restrict__ S, float* __restrict__ MX,
    int nE) {
  __shared__ __align__(16) _Float16 sW0[HID * DINP];
  __shared__ __align__(16) _Float16 sW1[HID * 64];
  __shared__ __align__(16) _Float16 sA[4][16 * DINP];
  __shared__ __align__(16) _Float16 sH[4][16 * 64];
  __shared__ float sB0[HID], sB1[HID];

  const int tid = threadIdx.x, lane = tid & 31, w = tid >> 5;
  stage_weights<DIN, DINP, 128>(W0, B0, W1, B1, sW0, sW1, sB0, sB1, tid);
  // Zero the K-pad (48..63) region of the per-wave H tile once.
  for (int i = lane; i < 16 * 16; i += 32) {
    int r = i >> 4, c = i & 15;
    sH[w][r * 64 + HID + c] = (_Float16)0.f;
  }
  __syncthreads();

  const int nT = (nE + 15) >> 4;
  const int tile = blockIdx.x * 4 + w;
  const bool active = tile < nT;
  const int tBase = (active ? tile : 0) * 16;

  // Build A tile in 8-element (16B) chunks; 48|48|48 never straddles a source.
  _Float16* A = sA[w];
  constexpr int CPR = DIN / 8;  // 18 chunks per row
  for (int ch = lane; ch < 16 * CPR; ch += 32) {
    const int r = ch / CPR, k = (ch - r * CPR) * 8;
    int m = tBase + r; if (m >= nE) m = nE - 1;
    h8 out;
    if (k < HID) {
      const float4* p = (const float4*)(X + row[m] * HID + k);
      out = cvt8(p[0], p[1]);
    } else if (k < 2 * HID) {
      const float4* p = (const float4*)(X + col[m] * HID + (k - HID));
      out = cvt8(p[0], p[1]);
    } else {
      out = *(const h8*)(Ein + m * HID + (k - 2 * HID));
    }
    *(h8*)(A + r * DINP + k) = out;
  }
  if constexpr (DINP > DIN) {
    constexpr int PCH = (DINP - DIN) / 8;  // pad chunks per row
    for (int ch = lane; ch < 16 * PCH; ch += 32) {
      const int r = ch / PCH, k = DIN + (ch - r * PCH) * 8;
      *(h8*)(A + r * DINP + k) = h8zero();
    }
  }
  __syncthreads();

  // Stage 1: H = relu(A @ W0^T + b0), 3 N-tiles x DINP/32 K-tiles.
  v8f c[3] = {};
#pragma unroll
  for (int kt = 0; kt < DINP / 32; ++kt) {
    v16h a = loadA(A, lane, kt, DINP);
#pragma unroll
    for (int nt = 0; nt < 3; ++nt)
      c[nt] = wmma_f16(a, loadB(sW0, lane, nt, kt, DINP), c[nt]);
  }
  const int half = lane >> 4, nl = lane & 15;
#pragma unroll
  for (int nt = 0; nt < 3; ++nt) {
    const int n = nt * 16 + nl;
    const float bb = sB0[n];
#pragma unroll
    for (int j = 0; j < 8; ++j) {
      float v = c[nt][j] + bb;
      sH[w][(half * 8 + j) * 64 + n] = (_Float16)(v > 0.f ? v : 0.f);
    }
  }
  __syncthreads();

  // Stage 2: E = H @ W1^T + b1, 3 N-tiles x 2 K-tiles.
  v8f d[3] = {};
#pragma unroll
  for (int kt = 0; kt < 2; ++kt) {
    v16h a = loadA(sH[w], lane, kt, 64);
#pragma unroll
    for (int nt = 0; nt < 3; ++nt)
      d[nt] = wmma_f16(a, loadB(sW1, lane, nt, kt, 64), d[nt]);
  }

  // Scatter atomics + stage E tile (f16, 16 x 48 row-major) back into sA.
  _Float16* Etile = sA[w];
  if (active) {
#pragma unroll
    for (int j = 0; j < 8; ++j) {
      const int m = tBase + half * 8 + j;
      if (m < nE) {
        const int ci = col[m];
#pragma unroll
        for (int nt = 0; nt < 3; ++nt) {
          const int n = nt * 16 + nl;
          const float v = d[nt][j] + sB1[n];
          Etile[(half * 8 + j) * HID + n] = (_Float16)v;
          atomicAdd(&S[ci * HID + n], v);
          atomicMaxF(&MX[ci * HID + n], v);
        }
      }
    }
  }
  __syncthreads();
  // Coalesced 16B stores of the E tile (3 chunks per lane).
  if (active) {
    for (int ch = lane; ch < 16 * 6; ch += 32) {
      const int r = ch / 6, k = (ch - r * 6) * 8;
      const int m = tBase + r;
      if (m < nE) *(h8*)(Eout + m * HID + k) = *(const h8*)(Etile + r * HID + k);
    }
  }
}

// ---------------------------------------------------------------------------
// Fused MLP2 via WMMA, node version. A = [x? | sum | max | mean].
// ---------------------------------------------------------------------------
template <int DIN, int DINP, bool HASX>
__global__ __launch_bounds__(128) void k_node_wmma(
    const float* __restrict__ X, const float* __restrict__ S,
    const float* __restrict__ MX, const float* __restrict__ CNT,
    const float* __restrict__ W0, const float* __restrict__ B0,
    const float* __restrict__ W1, const float* __restrict__ B1,
    float* __restrict__ Xout, int nN) {
  __shared__ __align__(16) _Float16 sW0[HID * DINP];
  __shared__ __align__(16) _Float16 sW1[HID * 64];
  __shared__ __align__(16) _Float16 sA[4][16 * DINP];
  __shared__ __align__(16) _Float16 sH[4][16 * 64];
  __shared__ float sB0[HID], sB1[HID];

  const int tid = threadIdx.x, lane = tid & 31, w = tid >> 5;
  stage_weights<DIN, DINP, 128>(W0, B0, W1, B1, sW0, sW1, sB0, sB1, tid);
  for (int i = lane; i < 16 * 16; i += 32) {
    int r = i >> 4, c = i & 15;
    sH[w][r * 64 + HID + c] = (_Float16)0.f;
  }
  __syncthreads();

  const int nT = (nN + 15) >> 4;
  const int tile = blockIdx.x * 4 + w;
  const bool active = tile < nT;
  const int tBase = (active ? tile : 0) * 16;

  _Float16* A = sA[w];
  constexpr int CPR = DIN / 8;  // 18 or 24 chunks per row
  for (int ch = lane; ch < 16 * CPR; ch += 32) {
    const int r = ch / CPR, k = (ch - r * CPR) * 8;
    int m = tBase + r; if (m >= nN) m = nN - 1;
    int kk = k;
    h8 out;
    bool handled = false;
    if constexpr (HASX) {
      if (k < HID) {
        const float4* p = (const float4*)(X + m * HID + k);
        out = cvt8(p[0], p[1]);
        handled = true;
      } else {
        kk = k - HID;
      }
    }
    if (!handled) {
      if (kk < HID) {                                       // sum
        const float4* p = (const float4*)(S + m * HID + kk);
        out = cvt8(p[0], p[1]);
      } else if (kk < 2 * HID) {                            // max (empty -> 0)
        const float cnt = CNT[m];
        const float4* p = (const float4*)(MX + m * HID + (kk - HID));
        out = (cnt > 0.f) ? cvt8(p[0], p[1]) : h8zero();
      } else {                                              // mean
        const float inv = 1.f / fmaxf(CNT[m], 1.f);
        const float4* p = (const float4*)(S + m * HID + (kk - 2 * HID));
        out = cvt8s(p[0], p[1], inv);
      }
    }
    *(h8*)(A + r * DINP + k) = out;
  }
  if constexpr (DINP > DIN) {
    constexpr int PCH = (DINP - DIN) / 8;
    for (int ch = lane; ch < 16 * PCH; ch += 32) {
      const int r = ch / PCH, k = DIN + (ch - r * PCH) * 8;
      *(h8*)(A + r * DINP + k) = h8zero();
    }
  }
  __syncthreads();

  v8f c[3] = {};
#pragma unroll
  for (int kt = 0; kt < DINP / 32; ++kt) {
    v16h a = loadA(A, lane, kt, DINP);
#pragma unroll
    for (int nt = 0; nt < 3; ++nt)
      c[nt] = wmma_f16(a, loadB(sW0, lane, nt, kt, DINP), c[nt]);
  }
  const int half = lane >> 4, nl = lane & 15;
#pragma unroll
  for (int nt = 0; nt < 3; ++nt) {
    const int n = nt * 16 + nl;
    const float bb = sB0[n];
#pragma unroll
    for (int j = 0; j < 8; ++j) {
      float v = c[nt][j] + bb;
      sH[w][(half * 8 + j) * 64 + n] = (_Float16)(v > 0.f ? v : 0.f);
    }
  }
  __syncthreads();

  v8f d[3] = {};
#pragma unroll
  for (int kt = 0; kt < 2; ++kt) {
    v16h a = loadA(sH[w], lane, kt, 64);
#pragma unroll
    for (int nt = 0; nt < 3; ++nt)
      d[nt] = wmma_f16(a, loadB(sW1, lane, nt, kt, 64), d[nt]);
  }

  // Stage X tile (f32, 16 x 48) into sA (reused as float buffer), then
  // emit coalesced float4 stores (6 per lane).
  float* Ftile = (float*)sA[w];
  if (active) {
#pragma unroll
    for (int nt = 0; nt < 3; ++nt) {
      const int n = nt * 16 + nl;
      const float bb = sB1[n];
#pragma unroll
      for (int j = 0; j < 8; ++j)
        Ftile[(half * 8 + j) * HID + n] = d[nt][j] + bb;
    }
  }
  __syncthreads();
  if (active) {
    for (int ch = lane; ch < 16 * 12; ch += 32) {
      const int r = ch / 12, k = (ch - r * 12) * 4;
      const int m = tBase + r;
      if (m < nN) *(float4*)(Xout + m * HID + k) = *(const float4*)(Ftile + r * HID + k);
    }
  }
}

// ---------------------------------------------------------------------------
// Per-graph pooling: batch is sorted -> block-local LDS reduction (ds atomics),
// then one global atomic per (graph, feature) per block.
// ---------------------------------------------------------------------------
__global__ __launch_bounds__(256) void k_pool(
    const float* __restrict__ X, const int* __restrict__ batch,
    float* __restrict__ GS, float* __restrict__ GMX, float* __restrict__ GCNT,
    int nN) {
  __shared__ float ls[16 * HID];
  __shared__ float lmx[16 * HID];
  __shared__ float lc[16];
  const int tid = threadIdx.x;
  for (int i = tid; i < 16 * HID; i += 256) { ls[i] = 0.f; lmx[i] = -FLT_MAX; }
  if (tid < 16) lc[tid] = 0.f;
  __syncthreads();

  const int base = blockIdx.x * 256;
  for (int idx = tid; idx < 256 * HID; idx += 256) {
    const int r = idx / HID, n = idx - r * HID;
    const int i = base + r;
    if (i < nN) {
      const int g = batch[i];
      const float v = X[i * HID + n];
      atomicAdd(&ls[g * HID + n], v);
      atomicMaxF(&lmx[g * HID + n], v);
      if (n == 0) atomicAdd(&lc[g], 1.f);
    }
  }
  __syncthreads();

  for (int i = tid; i < 16 * HID; i += 256) {
    if (ls[i] != 0.f) atomicAdd(&GS[i], ls[i]);
    if (lmx[i] > -FLT_MAX) atomicMaxF(&GMX[i], lmx[i]);
  }
  if (tid < 16 && lc[tid] > 0.f) atomicAdd(&GCNT[tid], lc[tid]);
}

// ---------------------------------------------------------------------------
// Head: 16 graphs, [gs|gmean|gmax](144) -> 48 -> 48 -> 48 -> 2. One lane/graph.
// ---------------------------------------------------------------------------
__global__ __launch_bounds__(32) void k_head(
    const float* __restrict__ GS, const float* __restrict__ GMX,
    const float* __restrict__ GCNT,
    const float* __restrict__ W0, const float* __restrict__ B0,
    const float* __restrict__ W1, const float* __restrict__ B1,
    const float* __restrict__ W2, const float* __restrict__ B2,
    const float* __restrict__ W3, const float* __restrict__ B3,
    float* __restrict__ out) {
  const int g = threadIdx.x;
  if (g >= 16) return;
  float a[3 * HID];
  const float c = GCNT[g];
  for (int n = 0; n < HID; ++n) {
    const float s = GS[g * HID + n];
    a[n] = s;
    a[HID + n] = s / fmaxf(c, 1.f);
    const float mx = GMX[g * HID + n];
    a[2 * HID + n] = (c > 0.f) ? mx : 0.f;
  }
  float h0[HID], h1[HID], h2[HID];
  for (int n = 0; n < HID; ++n) {
    float acc = B0[n];
    for (int k = 0; k < 3 * HID; ++k) acc += W0[n * 3 * HID + k] * a[k];
    h0[n] = acc > 0.f ? acc : 0.f;
  }
  for (int n = 0; n < HID; ++n) {
    float acc = B1[n];
    for (int k = 0; k < HID; ++k) acc += W1[n * HID + k] * h0[k];
    h1[n] = acc > 0.f ? acc : 0.f;
  }
  for (int n = 0; n < HID; ++n) {
    float acc = B2[n];
    for (int k = 0; k < HID; ++k) acc += W2[n * HID + k] * h1[k];
    h2[n] = acc > 0.f ? acc : 0.f;
  }
  for (int d = 0; d < 2; ++d) {
    float acc = B3[d];
    for (int k = 0; k < HID; ++k) acc += W3[d * HID + k] * h2[k];
    out[g * 2 + d] = acc;
  }
}

// ---------------------------------------------------------------------------
// Host-side orchestration.
// Input order (setup_inputs dict order, params flattened in insertion order):
//  0:x  1:edge_index(int32, [2,E])  2:edge_attr  3:batch(int32)
//  4..7  L0 edge  (W0 48x3,  b0, W1 48x48, b1)
//  8..11 L0 node  (W0 48x144,...)
//  12..15 L1 edge (48x144)   16..19 L1 node (48x192)
//  20..23 L2 edge            24..27 L2 node
//  28..35 out l0..l3
// ---------------------------------------------------------------------------
extern "C" void kernel_launch(void* const* d_in, const int* in_sizes, int n_in,
                              void* d_out, int out_size, void* d_ws, size_t ws_size,
                              hipStream_t stream) {
  (void)n_in; (void)out_size; (void)ws_size;
  const int* ei     = (const int*)d_in[1];
  const float* attr = (const float*)d_in[2];
  const int* batch  = (const int*)d_in[3];
  const int nE = in_sizes[2] / 3;
  const int nN = in_sizes[0] / 7;
  const int* row = ei;
  const int* col = ei + nE;
  auto W = [&](int i) { return (const float*)d_in[i]; };

  char* base = (char*)d_ws;
  size_t off = 0;
  auto take = [&](size_t bytes) -> char* {
    char* p = base + off;
    off = (off + bytes + 255) & ~(size_t)255;
    return p;
  };
  float*    CNT  = (float*)take((size_t)nN * 4);
  float*    S    = (float*)take((size_t)nN * HID * 4);
  float*    MX   = (float*)take((size_t)nN * HID * 4);
  float*    X0   = (float*)take((size_t)nN * HID * 4);
  float*    X1   = (float*)take((size_t)nN * HID * 4);
  _Float16* E0   = (_Float16*)take((size_t)nE * HID * 2);
  _Float16* E1   = (_Float16*)take((size_t)nE * HID * 2);
  float*    GS   = (float*)take(16 * HID * 4);
  float*    GMX  = (float*)take(16 * HID * 4);
  float*    GCNT = (float*)take(16 * 4);

  auto fill = [&](float* p, float v, int n) {
    k_fill<<<(n + 255) / 256, 256, 0, stream>>>(p, v, n);
  };

  // Degree counts (col never changes) + pool accumulator init.
  fill(CNT, 0.f, nN);
  k_count<<<(nE + 255) / 256, 256, 0, stream>>>(col, CNT, nE);
  fill(GS, 0.f, 16 * HID);
  fill(GMX, -FLT_MAX, 16 * HID);
  fill(GCNT, 0.f, 16);

  const int tE = (nE + 15) / 16, bE = (tE + 3) / 4;
  const int tN = (nN + 15) / 16, bN = (tN + 3) / 4;

  // ---- layer 0 ----
  fill(S, 0.f, nN * HID);
  fill(MX, -FLT_MAX, nN * HID);
  k_edge0<<<(nE + 255) / 256, 256, 0, stream>>>(attr, col, W(4), W(5), W(6), W(7),
                                                E0, S, MX, nE);
  k_node_wmma<144, 160, false><<<bN, 128, 0, stream>>>(
      nullptr, S, MX, CNT, W(8), W(9), W(10), W(11), X0, nN);

  // ---- layer 1 ----
  fill(S, 0.f, nN * HID);
  fill(MX, -FLT_MAX, nN * HID);
  k_edge_wmma<144, 160><<<bE, 128, 0, stream>>>(
      X0, E0, row, col, W(12), W(13), W(14), W(15), E1, S, MX, nE);
  k_node_wmma<192, 192, true><<<bN, 128, 0, stream>>>(
      X0, S, MX, CNT, W(16), W(17), W(18), W(19), X1, nN);

  // ---- layer 2 ----
  fill(S, 0.f, nN * HID);
  fill(MX, -FLT_MAX, nN * HID);
  k_edge_wmma<144, 160><<<bE, 128, 0, stream>>>(
      X1, E1, row, col, W(20), W(21), W(22), W(23), E0, S, MX, nE);
  k_node_wmma<192, 192, true><<<bN, 128, 0, stream>>>(
      X1, S, MX, CNT, W(24), W(25), W(26), W(27), X0, nN);

  // ---- pool + head ----
  k_pool<<<(nN + 255) / 256, 256, 0, stream>>>(X0, batch, GS, GMX, GCNT, nN);
  k_head<<<1, 32, 0, stream>>>(GS, GMX, GCNT, W(28), W(29), W(30), W(31),
                               W(32), W(33), W(34), W(35), (float*)d_out);
}
